// ConvLSTMIncr_17575006175560
// MI455X (gfx1250) — matrix operand order repaired and checked by reading, not compiled, not run on hardware
//
#include <hip/hip_runtime.h>
#include <stdint.h>

// ---------------------------------------------------------------------------
// ConvLSTM cell, implicit-GEMM via CDNA5 WMMA bf16 + async-to-LDS staging.
// B=8, Cin=Ch=64, H=W=128, K=3.
// GEMM: D[M=256][N=131072] = A[256][1152] * B[1152][N],  K=(ky,kx,c), c fastest.
// Activations packed NHWC bf16 with a zero halo: [B][130][130][128].
// ---------------------------------------------------------------------------

typedef __attribute__((ext_vector_type(16))) __bf16 v16bf;
typedef __attribute__((ext_vector_type(8)))  float  v8f;

union Frag16 { uint4 u[2]; v16bf v; };

#define HW    128
#define HP    130                        // padded H/W (1-px zero halo)
#define BATCH 8
#define CH    64
#define KDIM  1152                       // 128 * 9
#define NPIX  (BATCH * HW * HW)          // 131072
#define NPAD  (BATCH * HP * HP)          // 135200 padded pixels
#define HIDN  ((size_t)BATCH * CH * HW * HW)

__device__ __forceinline__ uint16_t f32_to_bf16(float f) {
  union { float f; uint32_t u; } c; c.f = f;
  uint32_t u = c.u;
  uint32_t r = u + 0x7FFFu + ((u >> 16) & 1u);   // round-to-nearest-even
  return (uint16_t)(r >> 16);
}

__device__ __forceinline__ float fsig(float v) {
  return __builtin_amdgcn_rcpf(1.0f + __expf(-v));
}
__device__ __forceinline__ float ftanh(float v) {
  return 1.0f - 2.0f * __builtin_amdgcn_rcpf(__expf(2.0f * v) + 1.0f);
}

// --------------------------------------------------------------------------
// Pack x | prev_h (B,64,H,W) fp32 -> padded NHWC bf16 [B,130,130,128], zero halo.
// --------------------------------------------------------------------------
__global__ void pack_stacked(const float* __restrict__ x,
                             const float* __restrict__ h,
                             uint16_t* __restrict__ dst) {
  int p = blockIdx.x * blockDim.x + threadIdx.x;   // padded pixel id
  if (p >= NPAD) return;
  int b  = p / (HP * HP);
  int rm = p - b * (HP * HP);
  int py = rm / HP;
  int px = rm - py * HP;
  uint16_t* d = dst + (size_t)p * 128;
  int y = py - 1, xx = px - 1;
  if (y < 0 || y >= HW || xx < 0 || xx >= HW) {
    uint4 z = make_uint4(0u, 0u, 0u, 0u);
#pragma unroll
    for (int c = 0; c < 8; ++c) ((uint4*)d)[c] = z;   // zero halo
    return;
  }
  const float* xs = x + (((size_t)b * CH) * HW + y) * HW + xx;
  const float* hs = h + (((size_t)b * CH) * HW + y) * HW + xx;
#pragma unroll 8
  for (int c = 0; c < 64; ++c) d[c]      = f32_to_bf16(xs[(size_t)c * HW * HW]);
#pragma unroll 8
  for (int c = 0; c < 64; ++c) d[64 + c] = f32_to_bf16(hs[(size_t)c * HW * HW]);
}

// --------------------------------------------------------------------------
// Pack W_gates (256,128,3,3) OIHW fp32 -> bf16 [256][1152], K=(ky,kx,c)
// --------------------------------------------------------------------------
__global__ void pack_weights(const float* __restrict__ w, uint16_t* __restrict__ dst) {
  int i = blockIdx.x * blockDim.x + threadIdx.x;
  if (i >= 256 * KDIM) return;
  int m = i / KDIM, k = i - m * KDIM;
  int slice = k >> 7, c = k & 127;
  int ky = slice / 3, kx = slice - ky * 3;
  dst[i] = f32_to_bf16(w[(((size_t)m * 128 + c) * 3 + ky) * 3 + kx]);
}

// --------------------------------------------------------------------------
// Fused conv-GEMM + LSTM gating.
//   grid.x = 2048 blocks, 256 threads (8 wave32).
//   Block: full M=256 (4 gates x 64 ch) x 64 pixels.
//   Wave w: channels [(w&3)*16, +16) for all four gates, pixel group w>>2
//   => each wave holds i/f/o/g for identical (ch,pixel) coords in registers.
// B tiles: GLOBAL_LOAD_ASYNC_TO_LDS_B128 (ASYNCcnt), double-buffered LDS.
// A fragments: register double-buffer with NAMED buffers (no dynamic index,
// no spill); K-loop manually unrolled by 2 so buffer selection is static.
// Barrier: raw s_wait_asynccnt + s_barrier (no loadcnt drain -> A loads
// issued at top of a step retire under that step's WMMAs).
// --------------------------------------------------------------------------
__global__ __launch_bounds__(256)
void convlstm_gemm(const uint16_t* __restrict__ Abf,     // [256][1152] bf16
                   const uint16_t* __restrict__ Sbf,     // padded NHWC bf16
                   const float*    __restrict__ bias,    // [256]
                   const float*    __restrict__ prev_c,  // [8][64][128][128]
                   float*          __restrict__ out) {   // hidden | cell
  // B tile: 64 px x 32 K bf16, pixel stride 40 u16 (80B). Double buffer = 10 KB.
  __shared__ __align__(16) uint16_t lds[2][64 * 40];

  const int tid    = threadIdx.x;
  const int lane   = tid & 31;
  const int lidx   = lane & 15;
  const int half   = lane >> 4;
  const int wave   = tid >> 5;
  const int g      = wave >> 2;            // pixel group 0/1
  const int chBase = (wave & 3) * 16;      // channel sub-tile within gate

  const int n0  = blockIdx.x * 64;         // row-aligned (128 % 64 == 0)
  const int b   = n0 >> 14;
  const int rem = n0 & 16383;
  const int y   = rem >> 7;
  const int x0  = rem & 127;

  const int lp = tid >> 2;                 // pixel 0..63 (cooperative loader)
  const int lc = (tid & 3) * 8;            // c-chunk within 32-K tile

  // per-thread constant part of the global B address (padded coords)
  const uint16_t* gbase = Sbf + (((size_t)b * HP + y) * HP + (x0 + lp)) * 128 + lc;

  // Async global->LDS copy of this thread's 16B chunk of B tile s.
  auto asyncB = [&](int s) {
    int slice = s >> 2;                    // (ky,kx) slice 0..8
    int c0    = (s & 3) * 32;
    int ky = slice / 3, kx = slice - ky * 3;
    uint64_t gaddr  = (uint64_t)(uintptr_t)(gbase + ((size_t)(ky * HP + kx)) * 128 + c0);
    uint32_t ldsoff = (uint32_t)(uintptr_t)(&lds[s & 1][lp * 40 + lc]);
    asm volatile("global_load_async_to_lds_b128 %0, %1, off"
                 :: "v"(ldsoff), "v"(gaddr) : "memory");
  };

  auto loadA = [&](int s, Frag16* af) {
    const size_t k0 = (size_t)s * 32;
#pragma unroll
    for (int q = 0; q < 4; ++q) {
      int m = q * 64 + chBase + lidx;
      const uint16_t* base = Abf + (size_t)m * KDIM + k0 + half * 8;
      af[q].u[0] = *(const uint4*)(base);        // K [half*8, +8)
      af[q].u[1] = *(const uint4*)(base + 16);   // K [16+half*8, +8)
    }
  };

  v8f acc[4][2] = {};                      // [gate][n-tile] fp32 accumulators

  // drain own asyncs, then split workgroup barrier (no vmem-load drain).
  auto wgbar = [&]() {
    asm volatile("s_wait_asynccnt 0x0\n\t"
                 "s_barrier_signal -1\n\t"
                 "s_barrier_wait -1" ::: "memory");
  };

  auto compute = [&](const uint16_t* buf, Frag16* a) {
    // B fragments (K=32 x N=16): lane = col lidx, K [half*8,+8) and [16+half*8,+8).
    Frag16 bf[2];
#pragma unroll
    for (int t = 0; t < 2; ++t) {
      int pix = g * 32 + t * 16 + lidx;
      const uint4* p = (const uint4*)(buf + pix * 40 + half * 8);
      bf[t].u[0] = p[0];
      bf[t].u[1] = p[2];
    }
#pragma unroll
    for (int q = 0; q < 4; ++q)
#pragma unroll
      for (int t = 0; t < 2; ++t)
        acc[q][t] = __builtin_amdgcn_wmma_f32_16x16x32_bf16(
            false, a[q].v, false, bf[t].v, (short)0, acc[q][t], false, false);
  };

  Frag16 afA[4], afB[4];                   // NAMED double buffers (stay in VGPRs)

  asyncB(0);
  wgbar();
  loadA(0, afA);

#pragma unroll 1
  for (int s = 0; s < 36; s += 2) {
    // ---- even step s: consume afA + lds[0]; prefetch s+1 into afB/lds[1]
    asyncB(s + 1);
    loadA(s + 1, afB);
    compute(&lds[0][0], afA);
    wgbar();

    // ---- odd step s+1: consume afB + lds[1]; prefetch s+2 into afA/lds[0]
    if (s + 2 < 36) {
      asyncB(s + 2);
      loadA(s + 2, afA);
    }
    compute(&lds[1][0], afB);
    wgbar();
  }

  // ---------------- fused LSTM gating epilogue -----------------------------
  // C/D layout: VGPR r, lanes 0-15: M=r, N=lane; lanes 16-31: M=8+r, N=lane-16.
#pragma unroll
  for (int t = 0; t < 2; ++t) {
    int xx = x0 + g * 32 + t * 16 + lidx;
#pragma unroll
    for (int r = 0; r < 8; ++r) {
      int chl = half * 8 + r;
      int ch  = chBase + chl;
      float vi = acc[0][t][r] + bias[ch];
      float vf = acc[1][t][r] + bias[64  + ch];
      float vo = acc[2][t][r] + bias[128 + ch];
      float vg = acc[3][t][r] + bias[192 + ch];
      float ig = fsig(vi);
      float fg = fsig(vf);
      float og = fsig(vo);
      float gg = ftanh(vg);
      size_t idx = (((size_t)b * CH + ch) * HW + y) * HW + xx;
      float pc   = prev_c[idx];
      float cell = ig * gg + fg * pc;
      float hid  = og * ftanh(cell);
      out[idx]        = hid;               // hidden
      out[HIDN + idx] = cell;              // cell
    }
  }
}

// --------------------------------------------------------------------------
extern "C" void kernel_launch(void* const* d_in, const int* in_sizes, int n_in,
                              void* d_out, int out_size, void* d_ws, size_t ws_size,
                              hipStream_t stream) {
  const float* x  = (const float*)d_in[0];
  const float* ph = (const float*)d_in[1];
  const float* pc = (const float*)d_in[2];
  const float* Wg = (const float*)d_in[3];
  const float* bg = (const float*)d_in[4];
  float* out = (float*)d_out;

  uint16_t* stacked = (uint16_t*)d_ws;                               // 34.6 MB
  uint16_t* wbf = (uint16_t*)((char*)d_ws + (size_t)NPAD * 128 * 2); // 0.59 MB

  pack_stacked<<<(NPAD + 255) / 256, 256, 0, stream>>>(x, ph, stacked);
  pack_weights<<<(256 * KDIM + 255) / 256, 256, 0, stream>>>(Wg, wbf);
  convlstm_gemm<<<NPIX / 64, 256, 0, stream>>>(wbf, stacked, bg, pc, out);
}
